// Model_77695958385158
// MI455X (gfx1250) — matrix operand order repaired and checked by reading, not compiled, not run on hardware
//
#include <hip/hip_runtime.h>
#include <hip/hip_bf16.h>

// ---------------------------------------------------------------------------
// BiLSTM + CRF Viterbi decode for MI455X (gfx1250, wave32, WMMA)
// B=128, S=512, V=30000, E=256, H=256, T=9
// ---------------------------------------------------------------------------

#define Bsz 128
#define Ssz 512
#define Esz 256
#define Hsz 256
#define Tsz 9
#define G4H 1024           // 4*H
#define Kc  512            // E + H (combined GEMM K)
#define BUF (Bsz * Kc)     // one LDS A-panel in u16 elements (128 KB)

typedef unsigned short u16;
typedef unsigned int   u32;
typedef unsigned long long u64;

typedef __attribute__((ext_vector_type(16))) __bf16 v16bf;
typedef __attribute__((ext_vector_type(8)))  float  v8f;

union V16 { uint4 q[2]; v16bf v; };

__device__ __forceinline__ float bf2f(u16 u) {
    u32 v = ((u32)u) << 16;
    float f;
    __builtin_memcpy(&f, &v, 4);
    return f;
}
__device__ __forceinline__ u16 f2bf(float f) {
    u32 v;
    __builtin_memcpy(&v, &f, 4);
    u32 r = (v + 0x7FFFu + ((v >> 16) & 1u)) >> 16;   // round-to-nearest-even
    return (u16)r;
}
__device__ __forceinline__ u32 pack2(float a, float b) {
    return (u32)f2bf(a) | ((u32)f2bf(b) << 16);
}
__device__ __forceinline__ float sigf(float x) {
    return 1.0f / (1.0f + __expf(-x));
}
// Branch-free tanh: no EXEC-mask branching on the recurrence critical path.
// |x| clamped so exp stays finite; accuracy ~1ulp of the exp path, plenty for
// LSTM gates feeding an argmax decode.
__device__ __forceinline__ float tanh_fast(float x) {
    float ax = fminf(__builtin_fabsf(x), 15.0f);
    float e  = __expf(2.0f * ax);
    float t  = 1.0f - 2.0f / (e + 1.0f);
    return __builtin_copysignf(t, x);
}

// ---------------------------------------------------------------------------
// Kernel 1: pack W_ih|W_hh (f32 row-major) -> combined bf16 [1024][512]
// ---------------------------------------------------------------------------
__global__ void prep_w_kernel(const float* __restrict__ Wih_f,
                              const float* __restrict__ Whh_f,
                              const float* __restrict__ Wih_b,
                              const float* __restrict__ Whh_b,
                              u16* __restrict__ Wf, u16* __restrict__ Wb) {
    int idx = blockIdx.x * blockDim.x + threadIdx.x;   // 0 .. 2*1024*512-1
    int dir = idx >> 19;
    int rem = idx & ((1 << 19) - 1);
    int n = rem >> 9;
    int k = rem & 511;
    const float* Wih = dir ? Wih_b : Wih_f;
    const float* Whh = dir ? Whh_b : Whh_f;
    float v = (k < Esz) ? Wih[n * Esz + k] : Whh[n * Hsz + (k - Esz)];
    (dir ? Wb : Wf)[rem] = f2bf(v);
}

// ---------------------------------------------------------------------------
// Kernel 2: embedding gather + f32->bf16.  Xbf layout: [(s*B+b)][E]
// ---------------------------------------------------------------------------
__global__ void embed_kernel(const int* __restrict__ ids,
                             const float* __restrict__ table,
                             u16* __restrict__ Xb) {
    int gid   = blockIdx.x * blockDim.x + threadIdx.x;  // tokens * 32
    int token = gid >> 5;                               // s*B + b
    int chunk = gid & 31;                               // 8 f32 per chunk
    int b = token & (Bsz - 1);
    int s = token >> 7;
    int id = ids[b * Ssz + s];
    const float4* src = (const float4*)(table + (size_t)id * Esz + chunk * 8);
    float4 f0 = src[0];
    float4 f1 = src[1];
    uint4 o;
    o.x = pack2(f0.x, f0.y);
    o.y = pack2(f0.z, f0.w);
    o.z = pack2(f1.x, f1.y);
    o.w = pack2(f1.z, f1.w);
    *(uint4*)(Xb + (size_t)token * Esz + chunk * 8) = o;
}

// ---------------------------------------------------------------------------
// Async x-panel loader: step s of X -> LDS buffer `dst` (cols 0..255), using
// gfx1250 GLOBAL_LOAD_ASYNC_TO_LDS_B128 (tracked by ASYNCcnt, not LOADcnt).
// Each of the 1024 threads moves 32 bf16 = 64 B = 4 x b128.
// ---------------------------------------------------------------------------
__device__ __forceinline__ void async_load_x(const u16* __restrict__ X,
                                             u16* dst, int s, int tid) {
    const int r  = tid >> 3;          // row 0..127
    const int ck = (tid & 7) * 32;    // 32 u16 per thread
    const u16* g = &X[((size_t)s * Bsz + r) * Esz + ck];
    u16* l = dst + r * Kc + ck;
    #pragma unroll
    for (int i = 0; i < 4; ++i) {
        u32 lds_addr = (u32)(u64)(l + i * 8);   // low 32 bits of LDS-aperture addr
        u64 gaddr    = (u64)(g + i * 8);
        asm volatile("global_load_async_to_lds_b128 %0, %1, off"
                     :: "v"(lds_addr), "v"(gaddr) : "memory");
    }
}
__device__ __forceinline__ void wait_async0() {
    asm volatile("s_wait_asynccnt 0" ::: "memory");
}

// ---------------------------------------------------------------------------
// Kernel 3: persistent LSTM recurrence, one workgroup per direction.
// Double-buffered LDS A-panels: [2][128][512] bf16 = 256 KB.
//   panel = [x_t (cols 0..255) | h_{t-1} (cols 256..511)]
// Per step: gates[128,1024] = A[128,512] x Wcomb^T via bf16 WMMA, fp32 acc.
// Wave (mhalf, ht) owns rows mhalf*64..+63 and hidden units ht*16..+15 across
// all four gate blocks, so each lane holds matching (i,f,g,o) quadruples and
// the LSTM cell state c lives in registers for all 512 steps.
// One barrier per step: h_t and x_{t+1} are written into the *other* panel.
// ---------------------------------------------------------------------------
__global__ void __launch_bounds__(1024, 1)
lstm_kernel(const u16* __restrict__ Wf, const u16* __restrict__ Wb,
            const u16* __restrict__ X,
            const float* __restrict__ bias_f, const float* __restrict__ bias_b,
            u16* __restrict__ Hf, u16* __restrict__ Hb) {
    extern __shared__ u16 smem[];                 // [2][128][512]
    const int dir = blockIdx.x;                   // 0 = fwd, 1 = bwd
    const u16*   W    = dir ? Wb : Wf;
    const float* bias = dir ? bias_b : bias_f;
    u16*         Hout = dir ? Hb : Hf;

    const int tid   = threadIdx.x;
    const int lane  = tid & 31;
    const int wid   = tid >> 5;
    const int mhalf = wid >> 4;      // 0..1  (rows mhalf*64 .. +63)
    const int ht    = wid & 15;      // hidden tile 0..15
    const int lmod  = lane & 15;
    const int lh    = lane >> 4;     // lane half

    // prologue: zero h region of panel 0, async-load x_0 into panel 0
    for (int i = tid; i < Bsz * Hsz; i += 1024) {
        int r = i >> 8, c = i & 255;
        smem[r * Kc + Esz + c] = 0;
    }
    async_load_x(X, smem, dir ? (Ssz - 1) : 0, tid);
    wait_async0();
    __syncthreads();

    float c_state[4][8];
    #pragma unroll
    for (int mt = 0; mt < 4; ++mt)
        #pragma unroll
        for (int j = 0; j < 8; ++j) c_state[mt][j] = 0.0f;

    float bgate[4];
    #pragma unroll
    for (int g = 0; g < 4; ++g) bgate[g] = bias[g * Hsz + ht * 16 + lmod];

    const v8f vzero = {0.f, 0.f, 0.f, 0.f, 0.f, 0.f, 0.f, 0.f};
    const int hid = ht * 16 + lmod;

    int cur = 0;
    for (int step = 0; step < Ssz; ++step) {
        const int s = dir ? (Ssz - 1 - step) : step;
        const u16* panel = smem + cur * BUF;
        u16*       nxt   = smem + (cur ^ 1) * BUF;

        // Two passes of 2 M-tiles each to cap accumulator register pressure.
        #pragma unroll
        for (int pass = 0; pass < 2; ++pass) {
            v8f acc[2][4];
            #pragma unroll
            for (int mi = 0; mi < 2; ++mi)
                #pragma unroll
                for (int g = 0; g < 4; ++g) acc[mi][g] = vzero;

            #pragma unroll 1   // keep tile loads inside the loop (no spills)
            for (int kc = 0; kc < 16; ++kc) {
                // A tiles from LDS (documented 16-bit A layout: two 8-elem K runs)
                V16 a[2];
                #pragma unroll
                for (int mi = 0; mi < 2; ++mi) {
                    int mt = pass * 2 + mi;
                    int r  = mhalf * 64 + mt * 16 + lmod;
                    const u16* p = &panel[r * Kc + kc * 32 + lh * 8];
                    a[mi].q[0] = *(const uint4*)p;          // K = kc*32+lh*8 .. +7
                    a[mi].q[1] = *(const uint4*)(p + 16);   // K = kc*32+16+lh*8 .. +7
                }
                // All four gate B-tiles up front -> one clause of 8 global
                // b128 loads, waits overlap with the first WMMAs.
                V16 bm[4];
                #pragma unroll
                for (int g = 0; g < 4; ++g) {
                    int n = g * Hsz + ht * 16 + lmod;
                    const u16* q = &W[(size_t)n * Kc + kc * 32 + lh * 8];
                    bm[g].q[0] = *(const uint4*)q;
                    bm[g].q[1] = *(const uint4*)(q + 16);
                }
                #pragma unroll
                for (int g = 0; g < 4; ++g)
                    #pragma unroll
                    for (int mi = 0; mi < 2; ++mi) {
                        acc[mi][g] = __builtin_amdgcn_wmma_f32_16x16x32_bf16(
                            false, a[mi].v, false, bm[g].v, (short)0, acc[mi][g],
                            false, false);
                    }
            }
            // LSTM elementwise: lane owns (m = mhalf*64+mt*16+lh*8+j, hid).
            // Branch-free gate math; h_t goes straight into the *next* panel
            // (no race: everyone reads `panel` this step) and to global H.
            #pragma unroll
            for (int mi = 0; mi < 2; ++mi) {
                int mt = pass * 2 + mi;
                #pragma unroll
                for (int j = 0; j < 8; ++j) {
                    float iv = acc[mi][0][j] + bgate[0];
                    float fv = acc[mi][1][j] + bgate[1];
                    float gv = acc[mi][2][j] + bgate[2];
                    float ov = acc[mi][3][j] + bgate[3];
                    float cn = sigf(fv) * c_state[mt][j] + sigf(iv) * tanh_fast(gv);
                    c_state[mt][j] = cn;
                    u16 hb = f2bf(sigf(ov) * tanh_fast(cn));
                    int m  = mhalf * 64 + mt * 16 + lh * 8 + j;
                    nxt[m * Kc + Esz + hid] = hb;
                    Hout[((size_t)s * Bsz + m) * Hsz + hid] = hb;
                }
            }
        }

        // prefetch x_{t+1} into the next panel via async-to-LDS
        if (step < Ssz - 1) {
            int ns = dir ? (Ssz - 2 - step) : (step + 1);
            async_load_x(X, nxt, ns, tid);
        }
        wait_async0();       // ASYNCcnt is not covered by __syncthreads
        __syncthreads();
        cur ^= 1;
    }
}

// ---------------------------------------------------------------------------
// Kernel 4: emissions  em[s*B+b][t] = b_out[t] + [hf|hb] . W_out[t]
// ---------------------------------------------------------------------------
__global__ void emissions_kernel(const u16* __restrict__ Hf,
                                 const u16* __restrict__ Hb,
                                 const float* __restrict__ W_out,
                                 const float* __restrict__ b_out,
                                 float* __restrict__ em) {
    __shared__ float w[Tsz * 2 * Hsz];
    __shared__ float bo[Tsz];
    for (int i = threadIdx.x; i < Tsz * 2 * Hsz; i += blockDim.x) w[i] = W_out[i];
    if (threadIdx.x < Tsz) bo[threadIdx.x] = b_out[threadIdx.x];
    __syncthreads();

    int p = blockIdx.x * blockDim.x + threadIdx.x;   // s*B + b
    float acc[Tsz];
    #pragma unroll
    for (int t = 0; t < Tsz; ++t) acc[t] = bo[t];

    const uint4* hf = (const uint4*)&Hf[(size_t)p * Hsz];
    const uint4* hb = (const uint4*)&Hb[(size_t)p * Hsz];
    for (int c = 0; c < Hsz / 8; ++c) {
        uint4 qf = hf[c];
        uint4 qb = hb[c];
        u32 wf[4]  = {qf.x, qf.y, qf.z, qf.w};
        u32 wb2[4] = {qb.x, qb.y, qb.z, qb.w};
        #pragma unroll
        for (int e = 0; e < 4; ++e) {
            int j0 = c * 8 + e * 2;
            float xf0 = bf2f((u16)(wf[e] & 0xFFFF));
            float xf1 = bf2f((u16)(wf[e] >> 16));
            float xb0 = bf2f((u16)(wb2[e] & 0xFFFF));
            float xb1 = bf2f((u16)(wb2[e] >> 16));
            #pragma unroll
            for (int t = 0; t < Tsz; ++t) {
                acc[t] += xf0 * w[t * 2 * Hsz + j0];
                acc[t] += xf1 * w[t * 2 * Hsz + j0 + 1];
                acc[t] += xb0 * w[t * 2 * Hsz + Hsz + j0];
                acc[t] += xb1 * w[t * 2 * Hsz + Hsz + j0 + 1];
            }
        }
    }
    #pragma unroll
    for (int t = 0; t < Tsz; ++t) em[(size_t)p * Tsz + t] = acc[t];
}

// ---------------------------------------------------------------------------
// Kernel 5: CRF Viterbi decode, one wave per batch element (lanes = tags)
// ---------------------------------------------------------------------------
__global__ void viterbi_kernel(const float* __restrict__ em,
                               const unsigned char* __restrict__ mask,
                               const float* __restrict__ start_t,
                               const float* __restrict__ end_t,
                               const float* __restrict__ trans,
                               unsigned char* __restrict__ hist,
                               int* __restrict__ out) {
    const int lane = threadIdx.x & 31;
    const int b    = blockIdx.x * (blockDim.x >> 5) + (threadIdx.x >> 5);
    const int j    = (lane < Tsz) ? lane : 0;

    float tcol[Tsz];
    #pragma unroll
    for (int i = 0; i < Tsz; ++i) tcol[i] = trans[i * Tsz + j];

    float score = start_t[j] + em[((size_t)0 * Bsz + b) * Tsz + j];

    for (int s = 1; s < Ssz; ++s) {
        float emv  = em[((size_t)s * Bsz + b) * Tsz + j];
        float best = -3.4e38f;
        int   bi   = 0;
        #pragma unroll
        for (int i = 0; i < Tsz; ++i) {
            float si = __shfl(score, i, 32) + tcol[i];
            if (si > best) { best = si; bi = i; }
        }
        hist[((size_t)(s - 1) * Bsz + b) * Tsz + j] = (unsigned char)bi;
        bool m = mask[b * Ssz + s] != 0;
        if (m) score = best + emv;
    }

    float total = score + end_t[j];
    int   last  = 0;
    float bv    = -3.4e38f;
    #pragma unroll
    for (int i = 0; i < Tsz; ++i) {
        float v = __shfl(total, i, 32);
        if (v > bv) { bv = v; last = i; }
    }

    if (lane == 0) {
        int tag = last;
        for (int s = Ssz - 1; s >= 1; --s) {
            bool m = mask[b * Ssz + s] != 0;
            out[b * Ssz + s] = m ? tag : 0;
            if (m) tag = hist[((size_t)(s - 1) * Bsz + b) * Tsz + tag];
        }
        out[b * Ssz + 0] = tag;   // position 0 always valid (length >= 1)
    }
}

// ---------------------------------------------------------------------------
// Host launcher
// ---------------------------------------------------------------------------
extern "C" void kernel_launch(void* const* d_in, const int* in_sizes, int n_in,
                              void* d_out, int out_size, void* d_ws, size_t ws_size,
                              hipStream_t stream) {
    const int*   input_ids = (const int*)d_in[0];
    const unsigned char* mask = (const unsigned char*)d_in[1]; // jnp bool -> 1 byte
    const float* embed   = (const float*)d_in[2];
    const float* W_ih_f  = (const float*)d_in[3];
    const float* W_hh_f  = (const float*)d_in[4];
    const float* b_f     = (const float*)d_in[5];
    const float* W_ih_b  = (const float*)d_in[6];
    const float* W_hh_b  = (const float*)d_in[7];
    const float* b_b     = (const float*)d_in[8];
    const float* W_out   = (const float*)d_in[9];
    const float* b_out   = (const float*)d_in[10];
    const float* start_t = (const float*)d_in[11];
    const float* end_t   = (const float*)d_in[12];
    const float* trans   = (const float*)d_in[13];
    int* out = (int*)d_out;

    char* ws = (char*)d_ws;
    const size_t MB = 1024 * 1024;
    u16*  Wf   = (u16*)(ws + 0);                    // 1 MB  bf16 [1024][512]
    u16*  Wb   = (u16*)(ws + 1 * MB);               // 1 MB
    u16*  Xbf  = (u16*)(ws + 2 * MB);               // 32 MB bf16 [(s*B+b)][256]
    u16*  Hf   = (u16*)(ws + 34 * MB);              // 32 MB bf16 [(s*B+b)][256]
    u16*  Hb   = (u16*)(ws + 66 * MB);              // 32 MB
    float* em  = (float*)(ws + 98 * MB);            // 2.25 MB f32 [(s*B+b)][9]
    unsigned char* hist = (unsigned char*)(ws + 98 * MB + 2359296); // 0.56 MB

    // 1) pack weights -> bf16 combined [4H][E+H]
    prep_w_kernel<<<4096, 256, 0, stream>>>(W_ih_f, W_hh_f, W_ih_b, W_hh_b, Wf, Wb);
    // 2) embedding gather -> bf16
    embed_kernel<<<8192, 256, 0, stream>>>(input_ids, embed, Xbf);
    // 3) BiLSTM recurrence (2 persistent workgroups, 256 KB dynamic LDS each)
    lstm_kernel<<<2, 1024, 2 * BUF * sizeof(u16), stream>>>(Wf, Wb, Xbf, b_f, b_b, Hf, Hb);
    // 4) emissions
    emissions_kernel<<<(Ssz * Bsz) / 256, 256, 0, stream>>>(Hf, Hb, W_out, b_out, em);
    // 5) Viterbi decode
    viterbi_kernel<<<Bsz / 4, 128, 0, stream>>>(em, mask, start_t, end_t, trans, hist, out);
}
